// enc_mtan_rnn_42855183679945
// MI455X (gfx1250) — matrix operand order repaired and checked by reading, not compiled, not run on hardware
//
#include <hip/hip_runtime.h>
#include <hip/hip_bf16.h>

// ---------------------------------------------------------------------------
// mTAN encoder for MI455X (gfx1250, wave32, WMMA).
// All GEMMs use v_wmma_f32_16x16x32_f16 (f16 A/B, f32 accum).
// Weights are pre-transposed to f16 "WT[n][k]" layout so B-fragments are
// two 16B-contiguous runs per lane.  Sequential GRU runs as 2 persistent workgroups
// (one per direction) with h + gates + recurrent weights resident in 160KB
// dynamic LDS, filled via gfx1250 async global->LDS copies.
// ---------------------------------------------------------------------------

typedef __attribute__((ext_vector_type(16))) _Float16 v16h;
typedef __attribute__((ext_vector_type(8)))  float    v8f;

#define BATCH 32
#define SEQL  128
#define SEQQ  128
#define IND   41
#define DIMC  82   // 2*IND
#define DPAD  96   // DIMC padded to 32
#define ET    128  // EMBED_TIME
#define NH    128  // NHID
#define G3    384  // 3*NHID
#define NL    50
#define NLP   64   // NLIN padded to 16
#define ROWS  4096 // BATCH*SEQQ

// ---- gfx1250 async global->LDS copy (ASYNCcnt-tracked) --------------------
__device__ inline void async_g2l_b128(unsigned lds_off, const void* gptr) {
  asm volatile("global_load_async_to_lds_b128 %0, %1, off"
               :: "v"(lds_off), "v"(gptr) : "memory");
}
__device__ inline void wait_asynccnt0() {
  asm volatile("s_wait_asynccnt 0x0" ::: "memory");
}

// ---- WMMA fragment loaders (layouts per CDNA5 ISA 7.12.2, wave32) ---------

// A-matrix 16x32 f16: lane m = lane&15; lanes>=16 use K-offset +8.
__device__ inline v16h load_a_frag(const _Float16* base, int stride) {
  int lane = threadIdx.x & 31;
  int m = lane & 15;
  int kbase = (lane >> 4) * 8;
  const _Float16* row = base + m * stride;
  v16h a;
#pragma unroll
  for (int i = 0; i < 8; ++i) {
    int k = (i < 4) ? (kbase + 2 * i) : (16 + kbase + 2 * (i - 4));
    a[2 * i]     = row[k];
    a[2 * i + 1] = row[k + 1];
  }
  return a;
}

// A fragment sourced from f32 data (GRU hidden state in LDS), converted.
__device__ inline v16h load_a_frag_f32(const float* base, int stride) {
  int lane = threadIdx.x & 31;
  int m = lane & 15;
  int kbase = (lane >> 4) * 8;
  const float* row = base + m * stride;
  v16h a;
#pragma unroll
  for (int i = 0; i < 8; ++i) {
    int k = (i < 4) ? (kbase + 2 * i) : (16 + kbase + 2 * (i - 4));
    a[2 * i]     = (_Float16)row[k];
    a[2 * i + 1] = (_Float16)row[k + 1];
  }
  return a;
}

// B-matrix 32x16 f16 from WT[n][k] storage (n = output col, k contiguous):
// lane n = lane&15, K = (lane>=16?8:0)+i for i<8, 16+(lane>=16?8:0)+(i-8) for i>=8.
__device__ inline v16h load_b_frag(const _Float16* base, int stride) {
  int lane = threadIdx.x & 31;
  int n = lane & 15;
  int kg = (lane >> 4) * 8;
  const _Float16* row = base + n * stride + kg;
  v16h b;
#pragma unroll
  for (int i = 0; i < 16; ++i) b[i] = row[(i < 8) ? i : (i + 8)];
  return b;
}

__device__ inline v8f wmma_f16(const v16h& a, const v16h& b, const v8f& c) {
  return __builtin_amdgcn_wmma_f32_16x16x32_f16(false, a, false, b, (short)0, c,
                                                false, false);
}

// ---- weight prep: f32 (K x N) -> f16 transposed+padded (Npad x Kpad) ------
__global__ void wt16_kernel(const float* __restrict__ src, _Float16* __restrict__ dst,
                            int K, int N, int Kpad, int Npad) {
  int idx = blockIdx.x * blockDim.x + threadIdx.x;
  if (idx >= Npad * Kpad) return;
  int n = idx / Kpad, k = idx % Kpad;
  float v = (n < N && k < K) ? src[k * N + n] : 0.0f;
  dst[idx] = (_Float16)v;
}

// ---- time embedding + Wq/Wk projection (block 0 = query, 1..32 = keys) ----
__global__ void embed_project_kernel(const float* __restrict__ tsteps,
                                     const float* __restrict__ query,
                                     const _Float16* __restrict__ WqT,
                                     const float* __restrict__ bq,
                                     const _Float16* __restrict__ WkT,
                                     const float* __restrict__ bk,
                                     _Float16* __restrict__ q_h,
                                     _Float16* __restrict__ k_h) {
  __shared__ _Float16 emb[128 * 128];
  int blk = blockIdx.x;
  const float* posv;
  const _Float16* WT;
  const float* bias;
  _Float16* outp;
  if (blk == 0) { posv = query;                    WT = WqT; bias = bq; outp = q_h; }
  else          { posv = tsteps + (blk - 1) * SEQL; WT = WkT; bias = bk;
                  outp = k_h + (size_t)(blk - 1) * SEQL * ET; }
  for (int i = threadIdx.x; i < 128 * 128; i += blockDim.x) {
    int row = i >> 7, c = i & 127;
    float pos = posv[row];
    float ex  = (float)(c & ~1) * (-2.302585093f / 128.0f); // arange(0,128,2)*(-ln10/128)
    float ang = 48.0f * pos * __expf(ex);
    float v   = (c & 1) ? __cosf(ang) : __sinf(ang);
    emb[i] = (_Float16)v;
  }
  __syncthreads();
  int mt = threadIdx.x >> 5;  // 8 waves -> 8 row tiles
  v16h a[4];
#pragma unroll
  for (int kc = 0; kc < 4; ++kc) a[kc] = load_a_frag(emb + mt * 16 * 128 + kc * 32, 128);
  int lane = threadIdx.x & 31;
  int mo = mt * 16 + ((lane >> 4) ? 8 : 0);
  for (int nt = 0; nt < 8; ++nt) {
    v8f acc = {};
#pragma unroll
    for (int kc = 0; kc < 4; ++kc)
      acc = wmma_f16(a[kc], load_b_frag(WT + nt * 16 * 128 + kc * 32, 128), acc);
    int n = nt * 16 + (lane & 15);
    float bv = bias[n];
#pragma unroll
    for (int r = 0; r < 8; ++r) outp[(mo + r) * ET + n] = (_Float16)(acc[r] + bv);
  }
}

// ---- scores[b][q][l] = (q . k^T)/sqrt(ET) ---------------------------------
__global__ void scores_kernel(const _Float16* __restrict__ q_h,
                              const _Float16* __restrict__ k_h,
                              float* __restrict__ scores) {
  int b = blockIdx.x;
  const _Float16* kb = k_h + (size_t)b * SEQL * ET;  // row l, col e  == B[e][l] source
  float* sb = scores + (size_t)b * SEQQ * SEQL;
  int mt = threadIdx.x >> 5;
  v16h a[4];
#pragma unroll
  for (int kc = 0; kc < 4; ++kc) a[kc] = load_a_frag(q_h + mt * 16 * ET + kc * 32, ET);
  int lane = threadIdx.x & 31;
  int mo = mt * 16 + ((lane >> 4) ? 8 : 0);
  const float inv = 0.08838834764831845f;  // 1/sqrt(128)
  for (int nt = 0; nt < 8; ++nt) {
    v8f acc = {};
#pragma unroll
    for (int kc = 0; kc < 4; ++kc)
      acc = wmma_f16(a[kc], load_b_frag(kb + nt * 16 * ET + kc * 32, ET), acc);
    int n = nt * 16 + (lane & 15);
#pragma unroll
    for (int r = 0; r < 8; ++r) sb[(mo + r) * SEQL + n] = acc[r] * inv;
  }
}

// ---- per-(q,d) masked softmax over L + weighted sum -> att (padded 96) ----
__global__ void softmax_att_kernel(const float* __restrict__ x,
                                   const float* __restrict__ scores,
                                   _Float16* __restrict__ att_h) {
  __shared__ __align__(16) float xs[SEQL * DIMC];  // 42KB: x[b] staged
  int b = blockIdx.x;
  const float* xb = x + (size_t)b * SEQL * DIMC;
  // async global->LDS stage of x[b]: 128*82*4 = 41984 B = 2624 x b128
  {
    unsigned base = (unsigned)(uintptr_t)&xs[0];
    const char* g = (const char*)xb;
    for (int i = threadIdx.x; i < (SEQL * DIMC * 4) / 16; i += blockDim.x)
      async_g2l_b128(base + i * 16, g + i * 16);
    wait_asynccnt0();
  }
  __syncthreads();
  const float* sb = scores + (size_t)b * SEQQ * SEQL;
  _Float16* ab = att_h + (size_t)b * SEQQ * DPAD;
  for (int p = threadIdx.x; p < SEQQ * DPAD; p += blockDim.x) {
    int qi = p / DPAD, d = p % DPAD;
    if (d >= DIMC) { ab[p] = (_Float16)0.0f; continue; }
    int mc = IND + (d % IND);  // mask column (mask duplicated over both halves)
    const float* srow = sb + qi * SEQL;
    float mx = -1e30f; bool any = false;
    for (int l = 0; l < SEQL; ++l)
      if (xs[l * DIMC + mc] != 0.0f) { mx = fmaxf(mx, srow[l]); any = true; }
    float sum = 0.f, acc = 0.f;
    if (any) {
      for (int l = 0; l < SEQL; ++l)
        if (xs[l * DIMC + mc] != 0.0f) {
          float e = __expf(srow[l] - mx);
          sum += e;
          acc += e * xs[l * DIMC + d];
        }
    }
    ab[p] = (_Float16)(any ? acc / sum : 0.0f);
  }
}

// ---- out = att @ Wo + bo  (K padded 82->96) -------------------------------
__global__ void outproj_kernel(const _Float16* __restrict__ att_h,
                               const _Float16* __restrict__ WoT,
                               const float* __restrict__ bo,
                               _Float16* __restrict__ out_h) {
  int b = blockIdx.x;
  const _Float16* ab = att_h + (size_t)b * SEQQ * DPAD;
  _Float16* ob = out_h + (size_t)b * SEQQ * NH;
  int mt = threadIdx.x >> 5;
  v16h a[3];
#pragma unroll
  for (int kc = 0; kc < 3; ++kc) a[kc] = load_a_frag(ab + mt * 16 * DPAD + kc * 32, DPAD);
  int lane = threadIdx.x & 31;
  int mo = mt * 16 + ((lane >> 4) ? 8 : 0);
  for (int nt = 0; nt < 8; ++nt) {
    v8f acc = {};
#pragma unroll
    for (int kc = 0; kc < 3; ++kc)
      acc = wmma_f16(a[kc], load_b_frag(WoT + nt * 16 * DPAD + kc * 32, DPAD), acc);
    int n = nt * 16 + (lane & 15);
    float bv = bo[n];
#pragma unroll
    for (int r = 0; r < 8; ++r) ob[(mo + r) * NH + n] = (_Float16)(acc[r] + bv);
  }
}

// ---- gx = out @ gWx + gbx  for both directions (grid.y = dir) -------------
__global__ void gx_kernel(const _Float16* __restrict__ out_h,
                          const _Float16* __restrict__ gWxTf, const float* __restrict__ gbxf,
                          const _Float16* __restrict__ gWxTb, const float* __restrict__ gbxb,
                          float* __restrict__ gx) {
  int mblk = blockIdx.x, dir = blockIdx.y;
  const _Float16* WT = dir ? gWxTb : gWxTf;
  const float* bias  = dir ? gbxb  : gbxf;
  const _Float16* ab = out_h + (size_t)mblk * 128 * NH;
  float* gb = gx + (size_t)dir * ROWS * G3 + (size_t)mblk * 128 * G3;
  int mt = threadIdx.x >> 5;
  v16h a[4];
#pragma unroll
  for (int kc = 0; kc < 4; ++kc) a[kc] = load_a_frag(ab + mt * 16 * NH + kc * 32, NH);
  int lane = threadIdx.x & 31;
  int mo = mt * 16 + ((lane >> 4) ? 8 : 0);
  for (int nt = 0; nt < 24; ++nt) {
    if (nt + 1 < 24) __builtin_prefetch(WT + (nt + 1) * 16 * NH, 0, 0);
    v8f acc = {};
#pragma unroll
    for (int kc = 0; kc < 4; ++kc)
      acc = wmma_f16(a[kc], load_b_frag(WT + nt * 16 * NH + kc * 32, NH), acc);
    int n = nt * 16 + (lane & 15);
    float bv = bias[n];
#pragma unroll
    for (int r = 0; r < 8; ++r) gb[(mo + r) * G3 + n] = acc[r] + bv;
  }
}

// ---- sequential GRU, one workgroup per direction --------------------------
// Dynamic LDS (160KB): h (16KB) | gh gates (48KB) | WhT f16 (96KB, async-copied)
__global__ void __launch_bounds__(512)
gru_kernel(const float* __restrict__ gx,
           const _Float16* __restrict__ gWhTf, const float* __restrict__ gbhf,
           const _Float16* __restrict__ gWhTb, const float* __restrict__ gbhb,
           _Float16* __restrict__ h16) {
  extern __shared__ float lds[];
  float* h  = lds;                                       // 32*128 f32
  float* gh = lds + BATCH * NH;                          // 32*384 f32
  _Float16* wt = (_Float16*)(lds + BATCH * NH + BATCH * G3);  // 384*128 f16
  int dir = blockIdx.x;
  const _Float16* WT = dir ? gWhTb : gWhTf;
  const float* bh    = dir ? gbhb  : gbhf;
  const float* g     = gx + (size_t)dir * ROWS * G3;
  // async-stage the recurrent weights into LDS: 384*128*2 = 98304 B = 6144 x b128
  {
    unsigned base = (unsigned)(uintptr_t)wt;
    const char* gsrc = (const char*)WT;
    for (int i = threadIdx.x; i < (G3 * NH * 2) / 16; i += blockDim.x)
      async_g2l_b128(base + i * 16, gsrc + i * 16);
  }
  for (int i = threadIdx.x; i < BATCH * NH; i += blockDim.x) h[i] = 0.f;
  wait_asynccnt0();
  __syncthreads();
  int wave = threadIdx.x >> 5;
  int lane = threadIdx.x & 31;
  for (int step = 0; step < SEQQ; ++step) {
    int t = dir ? (SEQQ - 1 - step) : step;
    // gh = h @ Wh : M=32 N=384 K=128 -> 48 tiles over 16 waves (3 each)
    for (int tile = wave * 3; tile < wave * 3 + 3; ++tile) {
      int mt = tile / 24, nt = tile % 24;
      v8f acc = {};
#pragma unroll
      for (int kc = 0; kc < 4; ++kc) {
        v16h a = load_a_frag_f32(h + mt * 16 * NH + kc * 32, NH);
        acc = wmma_f16(a, load_b_frag(wt + nt * 16 * NH + kc * 32, NH), acc);
      }
      int n = nt * 16 + (lane & 15);
      int mo = mt * 16 + ((lane >> 4) ? 8 : 0);
#pragma unroll
      for (int r = 0; r < 8; ++r) gh[(mo + r) * G3 + n] = acc[r];
    }
    __syncthreads();
    // gate update (4096 elements over 512 threads)
    for (int e = threadIdx.x; e < BATCH * NH; e += blockDim.x) {
      int bb = e >> 7, hid = e & 127;
      const float* gr = g + (size_t)(bb * SEQQ + t) * G3;
      float xr = gr[hid], xz = gr[NH + hid], xn = gr[2 * NH + hid];
      float hr = gh[bb * G3 + hid]          + bh[hid];
      float hz = gh[bb * G3 + NH + hid]     + bh[NH + hid];
      float hn = gh[bb * G3 + 2 * NH + hid] + bh[2 * NH + hid];
      float r  = 1.f / (1.f + __expf(-(xr + hr)));
      float zz = 1.f / (1.f + __expf(-(xz + hz)));
      float nn = tanhf(xn + r * hn);
      float hnew = (1.f - zz) * nn + zz * h[e];
      h[e] = hnew;
      h16[(size_t)(bb * SEQQ + t) * (2 * NH) + dir * NH + hid] = (_Float16)hnew;
    }
    __syncthreads();
  }
}

// ---- hidden = relu(h @ W1 + b1)  (N padded 50->64) ------------------------
__global__ void mlp1_kernel(const _Float16* __restrict__ h16,
                            const _Float16* __restrict__ W1T,
                            const float* __restrict__ b1,
                            _Float16* __restrict__ hid16) {
  int mblk = blockIdx.x;
  const _Float16* ab = h16 + (size_t)mblk * 128 * (2 * NH);
  _Float16* ob = hid16 + (size_t)mblk * 128 * NLP;
  int mt = threadIdx.x >> 5;
  v16h a[8];
#pragma unroll
  for (int kc = 0; kc < 8; ++kc)
    a[kc] = load_a_frag(ab + mt * 16 * (2 * NH) + kc * 32, 2 * NH);
  int lane = threadIdx.x & 31;
  int mo = mt * 16 + ((lane >> 4) ? 8 : 0);
  for (int nt = 0; nt < 4; ++nt) {
    v8f acc = {};
#pragma unroll
    for (int kc = 0; kc < 8; ++kc)
      acc = wmma_f16(a[kc], load_b_frag(W1T + nt * 16 * (2 * NH) + kc * 32, 2 * NH), acc);
    int n = nt * 16 + (lane & 15);
    float bv = (n < NL) ? b1[n] : 0.0f;
#pragma unroll
    for (int r = 0; r < 8; ++r) {
      float v = acc[r] + bv;
      ob[(mo + r) * NLP + n] = (_Float16)(v > 0.f ? v : 0.f);
    }
  }
}

// ---- z = hidden @ W2 + b2  (tiny: scalar) ---------------------------------
__global__ void mlp2_kernel(const _Float16* __restrict__ hid16,
                            const float* __restrict__ W2,
                            const float* __restrict__ b2,
                            float* __restrict__ z) {
  int idx = blockIdx.x * blockDim.x + threadIdx.x;
  if (idx >= ROWS * 40) return;
  int row = idx / 40, o = idx % 40;
  const _Float16* hr = hid16 + (size_t)row * NLP;
  float acc = b2[o];
#pragma unroll 10
  for (int j = 0; j < NL; ++j) acc += (float)hr[j] * W2[j * 40 + o];
  z[idx] = acc;
}

// ---------------------------------------------------------------------------
extern "C" void kernel_launch(void* const* d_in, const int* in_sizes, int n_in,
                              void* d_out, int out_size, void* d_ws, size_t ws_size,
                              hipStream_t stream) {
  const float* x      = (const float*)d_in[0];
  const float* tsteps = (const float*)d_in[1];
  const float* query  = (const float*)d_in[2];
  const float* Wq     = (const float*)d_in[3];
  const float* bq     = (const float*)d_in[4];
  const float* Wk     = (const float*)d_in[5];
  const float* bk     = (const float*)d_in[6];
  const float* Wo     = (const float*)d_in[7];
  const float* bo     = (const float*)d_in[8];
  const float* gWx_f  = (const float*)d_in[9];
  const float* gWh_f  = (const float*)d_in[10];
  const float* gbx_f  = (const float*)d_in[11];
  const float* gbh_f  = (const float*)d_in[12];
  const float* gWx_b  = (const float*)d_in[13];
  const float* gWh_b  = (const float*)d_in[14];
  const float* gbx_b  = (const float*)d_in[15];
  const float* gbh_b  = (const float*)d_in[16];
  const float* W1     = (const float*)d_in[17];
  const float* b1     = (const float*)d_in[18];
  const float* W2     = (const float*)d_in[19];
  const float* b2     = (const float*)d_in[20];
  float* z = (float*)d_out;

  // workspace carve-up (~21 MB total)
  char* ws = (char*)d_ws;
  size_t off = 0;
  auto carve = [&](size_t bytes) {
    char* p = ws + off;
    off = (off + bytes + 255) & ~(size_t)255;
    return p;
  };
  _Float16* WqT   = (_Float16*)carve(128 * 128 * 2);
  _Float16* WkT   = (_Float16*)carve(128 * 128 * 2);
  _Float16* WoT   = (_Float16*)carve(128 * DPAD * 2);
  _Float16* gWxTf = (_Float16*)carve(G3 * 128 * 2);
  _Float16* gWhTf = (_Float16*)carve(G3 * 128 * 2);
  _Float16* gWxTb = (_Float16*)carve(G3 * 128 * 2);
  _Float16* gWhTb = (_Float16*)carve(G3 * 128 * 2);
  _Float16* W1T   = (_Float16*)carve(NLP * 256 * 2);
  _Float16* q_h   = (_Float16*)carve((size_t)SEQQ * ET * 2);
  _Float16* k_h   = (_Float16*)carve((size_t)BATCH * SEQL * ET * 2);
  float*    scr   = (float*)   carve((size_t)BATCH * SEQQ * SEQL * 4);
  _Float16* att_h = (_Float16*)carve((size_t)BATCH * SEQQ * DPAD * 2);
  _Float16* out_h = (_Float16*)carve((size_t)BATCH * SEQQ * NH * 2);
  float*    gxws  = (float*)   carve((size_t)2 * ROWS * G3 * 4);
  _Float16* h16   = (_Float16*)carve((size_t)ROWS * 2 * NH * 2);
  _Float16* hid16 = (_Float16*)carve((size_t)ROWS * NLP * 2);

  // 1) weight prep (transpose+pad to f16)
  auto tl = [&](const float* s, _Float16* d, int K, int N, int Kp, int Np) {
    wt16_kernel<<<(Np * Kp + 255) / 256, 256, 0, stream>>>(s, d, K, N, Kp, Np);
  };
  tl(Wq, WqT, 128, 128, 128, 128);
  tl(Wk, WkT, 128, 128, 128, 128);
  tl(Wo, WoT, DIMC, 128, DPAD, 128);
  tl(gWx_f, gWxTf, 128, G3, 128, G3);
  tl(gWh_f, gWhTf, 128, G3, 128, G3);
  tl(gWx_b, gWxTb, 128, G3, 128, G3);
  tl(gWh_b, gWhTb, 128, G3, 128, G3);
  tl(W1, W1T, 256, NL, 256, NLP);

  // 2) time embedding + q/k projection (block 0 = query, 1..32 = per-batch keys)
  embed_project_kernel<<<BATCH + 1, 256, 0, stream>>>(tsteps, query, WqT, bq, WkT, bk, q_h, k_h);
  // 3) attention scores
  scores_kernel<<<BATCH, 256, 0, stream>>>(q_h, k_h, scr);
  // 4) masked softmax + weighted sum -> att (padded to 96)
  softmax_att_kernel<<<BATCH, 256, 0, stream>>>(x, scr, att_h);
  // 5) out = att @ Wo + bo
  outproj_kernel<<<BATCH, 256, 0, stream>>>(att_h, WoT, bo, out_h);
  // 6) gx = out @ gWx + gbx for both directions
  gx_kernel<<<dim3(32, 2), 256, 0, stream>>>(out_h, gWxTf, gbx_f, gWxTb, gbx_b, gxws);
  // 7) sequential bidirectional GRU (2 persistent WGs, 160KB dynamic LDS each)
  size_t gru_lds = (size_t)(BATCH * NH + BATCH * G3) * sizeof(float)
                 + (size_t)G3 * NH * sizeof(_Float16);
  gru_kernel<<<2, 512, gru_lds, stream>>>(gxws, gWhTf, gbh_f, gWhTb, gbh_b, h16);
  // 8) hidden = relu(h @ W1 + b1)
  mlp1_kernel<<<32, 256, 0, stream>>>(h16, W1T, b1, hid16);
  // 9) z = hidden @ W2 + b2
  mlp2_kernel<<<(ROWS * 40 + 255) / 256, 256, 0, stream>>>(hid16, W2, b2, z);
}
